// SDHGNN_89739046683506
// MI455X (gfx1250) — matrix operand order repaired and checked by reading, not compiled, not run on hardware
//
#include <hip/hip_runtime.h>
#include <hip/hip_bf16.h>
#include <stdint.h>

// ---------------- problem constants (from reference) ----------------
#define BQ   2      // batch
#define DCH  128    // channels D
#define NPT  2048   // points N
#define NHD  4      // heads
#define DHD  32     // head dim == WMMA K for bf16 (16x16x32)

typedef __attribute__((ext_vector_type(16))) __bf16 v16bf;
typedef __attribute__((ext_vector_type(8)))  float  v8f;

union FragU { v16bf v; uint4 q[2]; };

static __device__ __forceinline__ v8f wmma_bf(v16bf a, v16bf b, v8f c) {
  // D = A(16x32 bf16) * B(32x16 bf16) + C(16x16 f32)
  return __builtin_amdgcn_wmma_f32_16x16x32_bf16(false, a, false, b, (short)0, c, false, false);
}

// A fragment (16x32): source row-major, K contiguous, row stride rs (elements).
// lane: row = lane&15, g = lane>>4. Chunks: K = g*8+[0..7] and K = 16+g*8+[0..7]
// -> two 16B loads (global_load_b128 / ds_load_b128).
static __device__ __forceinline__ v16bf load_a_rm(const __bf16* base, int rs) {
  const int lane = threadIdx.x & 31;
  const int row  = lane & 15;
  const int g    = lane >> 4;
  const __bf16* p = base + (size_t)row * rs + g * 8;
  FragU f;
  f.q[0] = *(const uint4*)(p);
  f.q[1] = *(const uint4*)(p + 16);
  return f.v;
}

// B fragment (32x16): source stored K-contiguous per column, column stride cs (elements).
// lane: col = lane&15, g = lane>>4. Chunks: K = g*8+[0..7] and K = 16+g*8+[0..7] -> 2x b128.
static __device__ __forceinline__ v16bf load_b_kc(const __bf16* base, int cs) {
  const int lane = threadIdx.x & 31;
  const int col  = lane & 15;
  const int g    = lane >> 4;
  const __bf16* p = base + (size_t)col * cs + g * 8;
  FragU f;
  f.q[0] = *(const uint4*)(p);
  f.q[1] = *(const uint4*)(p + 16);
  return f.v;
}

// C/D 16x16 f32 layout: lane holds col = lane&15, rows m = (lane>>4)*8 + r

// ---------------- utility kernels ----------------
__global__ void k_cvt_bf16(const float* __restrict__ in, __bf16* __restrict__ out, int n) {
  int t = blockIdx.x * blockDim.x + threadIdx.x;
  if (t < n) out[t] = (__bf16)in[t];
}
__global__ void k_copy_f32(const float* __restrict__ in, float* __restrict__ out, int n) {
  int t = blockIdx.x * blockDim.x + threadIdx.x;
  if (t < n) out[t] = in[t];
}
// convert x [B,128,N] f32 (channel-major) -> xt [B,N,128] bf16 (point-major)
__global__ void k_cvt_xpose(const float* __restrict__ in, __bf16* __restrict__ out) {
  int t = blockIdx.x * blockDim.x + threadIdx.x;
  if (t >= BQ * DCH * NPT) return;
  int n = t % NPT;
  int c = (t / NPT) % DCH;
  int b = t / (NPT * DCH);
  out[((size_t)b * NPT + n) * DCH + c] = (__bf16)in[t];
}
__global__ void k_iota_idx(int* idx) { // [BQ,2048]
  int t = blockIdx.x * blockDim.x + threadIdx.x;
  if (t < BQ * NPT) idx[t] = t & (NPT - 1);
}
// row gather (point-major): st[b,j,:] = xt[b,idx[b,j],:]  (128 bf16 = 16 uint4, coalesced)
__global__ void k_gather_rows(const __bf16* __restrict__ xt, const int* __restrict__ idx,
                              __bf16* __restrict__ st, int M) {
  int t = blockIdx.x * blockDim.x + threadIdx.x;
  if (t >= BQ * M * 16) return;
  int qq = t & 15;
  int j  = (t >> 4) % M;
  int b  = t / (M * 16);
  const uint4* src = (const uint4*)(xt + ((size_t)b * NPT + idx[b * NPT + j]) * DCH);
  uint4* dst = (uint4*)(st + ((size_t)b * M + j) * DCH);
  dst[qq] = src[qq];
}
// copy xt rows into first 128 channels of y [B,N,256] (point-major concat)
__global__ void k_x_to_y(const __bf16* __restrict__ xt, __bf16* __restrict__ y) {
  int t = blockIdx.x * blockDim.x + threadIdx.x;
  if (t >= BQ * NPT * 16) return;
  int qq = t & 15;
  int n  = (t >> 4) % NPT;
  int b  = t / (NPT * 16);
  const uint4* src = (const uint4*)(xt + ((size_t)b * NPT + n) * DCH);
  uint4* dst = (uint4*)(y + ((size_t)b * NPT + n) * 256);
  dst[qq] = src[qq];
}

// ---------------- GEMM family: out[b,o,n] = sum_i W[o,i]*Xt[b,n,i] ----------------
static __device__ __forceinline__ bool wave_tile(int Cout, int NC, int& b, int& o0, int& n0) {
  int w = blockIdx.x * (blockDim.x >> 5) + (threadIdx.x >> 5);
  const int nt = NC >> 6, ot = Cout >> 4;
  if (w >= BQ * ot * nt) return false;
  n0 = (w % nt) << 6; w /= nt;
  o0 = (w % ot) << 4; w /= ot;
  b = w;
  return true;
}
// W row-major [Cout,Cin]; Xt point-major [NC,Cin] (batch base passed in)
static __device__ __forceinline__ void gemm_accum(const __bf16* __restrict__ Wrow,
                                                  const __bf16* __restrict__ Xt,
                                                  int Cin, int n0, v8f acc[4]) {
  for (int i0 = 0; i0 < Cin; i0 += 32) {
    v16bf a = load_a_rm(Wrow + i0, Cin);
#pragma unroll
    for (int j = 0; j < 4; ++j) {
      v16bf bf = load_b_kc(Xt + (size_t)(n0 + 16 * j) * Cin + i0, Cin);
      acc[j] = wmma_bf(a, bf, acc[j]);
    }
  }
}

// q/k/v projections. ptmajor=1: out[b,h,n,d] (q,k); ptmajor=0: out[b,h,d,n] (v)
__global__ void k_gemm_qkv(const __bf16* __restrict__ W, const float* __restrict__ bias,
                           const __bf16* __restrict__ Xt, __bf16* __restrict__ out,
                           int NC, int ptmajor) {
  int b, o0, n0;
  if (!wave_tile(DCH, NC, b, o0, n0)) return;
  v8f acc[4] = {};
  gemm_accum(W + (size_t)o0 * DCH, Xt + (size_t)b * NC * DCH, DCH, n0, acc);
  const int lane = threadIdx.x & 31, col = lane & 15, hs = lane >> 4;
#pragma unroll
  for (int j = 0; j < 4; ++j)
#pragma unroll
    for (int r = 0; r < 8; ++r) {
      int o = o0 + hs * 8 + r;
      int h = o & 3, d = o >> 2;
      int n = n0 + 16 * j + col;
      float val = acc[j][r] + bias[o];
      size_t bh = (size_t)(b * NHD + h);
      if (ptmajor) out[(bh * NC + n) * DHD + d] = (__bf16)val;
      else         out[(bh * DHD + d) * NC + n] = (__bf16)val;
    }
}
// bf16 output, point-major rows of stride RS: out[b*obst + n*RS + o] (Wm -> y[:,128:256])
__global__ void k_gemm_bf16t(const __bf16* __restrict__ W, const float* __restrict__ bias,
                             const __bf16* __restrict__ Xt, __bf16* __restrict__ out,
                             int Cin, int Cout, int NC, int RS, int obst) {
  int b, o0, n0;
  if (!wave_tile(Cout, NC, b, o0, n0)) return;
  v8f acc[4] = {};
  gemm_accum(W + (size_t)o0 * Cin, Xt + (size_t)b * NC * Cin, Cin, n0, acc);
  const int lane = threadIdx.x & 31, col = lane & 15, hs = lane >> 4;
#pragma unroll
  for (int j = 0; j < 4; ++j)
#pragma unroll
    for (int r = 0; r < 8; ++r) {
      int o = o0 + hs * 8 + r;
      int n = n0 + 16 * j + col;
      out[(size_t)b * obst + (size_t)n * RS + o] = (__bf16)(acc[j][r] + bias[o]);
    }
}
// fp32 channel-major output (W1 -> pre-instnorm tensor [B,256,N])
__global__ void k_gemm_f32(const __bf16* __restrict__ W, const float* __restrict__ bias,
                           const __bf16* __restrict__ Xt, float* __restrict__ out,
                           int Cin, int Cout, int NC) {
  int b, o0, n0;
  if (!wave_tile(Cout, NC, b, o0, n0)) return;
  v8f acc[4] = {};
  gemm_accum(W + (size_t)o0 * Cin, Xt + (size_t)b * NC * Cin, Cin, n0, acc);
  const int lane = threadIdx.x & 31, col = lane & 15, hs = lane >> 4;
#pragma unroll
  for (int j = 0; j < 4; ++j)
#pragma unroll
    for (int r = 0; r < 8; ++r) {
      int o = o0 + hs * 8 + r;
      out[(size_t)b * Cout * NC + (size_t)o * NC + n0 + 16 * j + col] = acc[j][r] + bias[o];
    }
}
// W2 + bias + residual: xf(channel-major) updated in place; xt(point-major bf16) refreshed
__global__ void k_gemm_resid(const __bf16* __restrict__ W, const float* __restrict__ bias,
                             const __bf16* __restrict__ Ht, float* __restrict__ xf,
                             __bf16* __restrict__ xt, int NC) {
  int b, o0, n0;
  if (!wave_tile(DCH, NC, b, o0, n0)) return;
  v8f acc[4] = {};
  gemm_accum(W + (size_t)o0 * 256, Ht + (size_t)b * NC * 256, 256, n0, acc);
  const int lane = threadIdx.x & 31, col = lane & 15, hs = lane >> 4;
#pragma unroll
  for (int j = 0; j < 4; ++j)
#pragma unroll
    for (int r = 0; r < 8; ++r) {
      int o = o0 + hs * 8 + r;
      int n = n0 + 16 * j + col;
      size_t ix = (size_t)b * DCH * NC + (size_t)o * NC + n;
      float val = acc[j][r] + bias[o] + xf[ix];
      xf[ix] = val;
      xt[((size_t)b * NC + n) * DCH + o] = (__bf16)val;
    }
}

// ---------------- attention pass 1: online-softmax row stats ----------------
// q:[B,H,N,32] k:[B,H,M,32] (point-major). rowmax/rowsum: [B*H, N] f32.
__global__ void k_attn_stats(const __bf16* __restrict__ q, const __bf16* __restrict__ kk,
                             float* __restrict__ rowmax, float* __restrict__ rowsum,
                             int M, float scale) {
  int w = blockIdx.x * (blockDim.x >> 5) + (threadIdx.x >> 5);
  const int nt = NPT >> 4;
  int n0 = (w % nt) << 4;
  int bh = w / nt;
  if (bh >= BQ * NHD) return;
  const __bf16* qb = q  + (size_t)bh * NPT * DHD;
  const __bf16* kb = kk + (size_t)bh * M * DHD;
  v16bf aq = load_a_rm(qb + (size_t)n0 * DHD, DHD);   // A[n][d], d contiguous
  const int lane = threadIdx.x & 31;
  float mx[8], ls[8];
#pragma unroll
  for (int r = 0; r < 8; ++r) { mx[r] = -3.0e38f; ls[r] = 0.f; }
  for (int m0 = 0; m0 < M; m0 += 16) {
    v16bf bk = load_b_kc(kb + (size_t)m0 * DHD, DHD); // B[d][m], d contiguous per column m
    v8f s = {};
    s = wmma_bf(aq, bk, s);
#pragma unroll
    for (int r = 0; r < 8; ++r) {
      float val = s[r] * scale;
      if (val > mx[r]) { ls[r] = ls[r] * __expf(mx[r] - val) + 1.0f; mx[r] = val; }
      else             { ls[r] += __expf(val - mx[r]); }
    }
  }
  // merge the 16 lane-columns of each half (rows differ between halves, xor<=8 stays in half)
#pragma unroll
  for (int off = 1; off < 16; off <<= 1) {
#pragma unroll
    for (int r = 0; r < 8; ++r) {
      float om = __shfl_xor(mx[r], off, 32);
      float ol = __shfl_xor(ls[r], off, 32);
      float nm = fmaxf(mx[r], om);
      ls[r] = ls[r] * __expf(mx[r] - nm) + ol * __expf(om - nm);
      mx[r] = nm;
    }
  }
  if ((lane & 15) == 0) {
    const int hs = lane >> 4;
#pragma unroll
    for (int r = 0; r < 8; ++r) {
      int n = n0 + hs * 8 + r;
      rowmax[(size_t)bh * NPT + n] = mx[r];
      rowsum[(size_t)bh * NPT + n] = ls[r];
    }
  }
}

// ---------------- attention pass 2: prob * v via WMMA + colsum partials ----------------
// v: [B,H,32,M] channel-major (m contiguous -> B-fragment friendly for K=m)
__global__ void k_attn_apply(const __bf16* __restrict__ q, const __bf16* __restrict__ kk,
                             const __bf16* __restrict__ vv, const float* __restrict__ rowmax,
                             const float* __restrict__ rowsum, float* __restrict__ pbuf,
                             __bf16* __restrict__ msgt, int M, float scale) {
  __shared__ __bf16 lds[8 * 16 * 32];
  const int wid = threadIdx.x >> 5;
  __bf16* pl = lds + wid * 512;
  int w = blockIdx.x * (blockDim.x >> 5) + wid;
  const int nt = NPT >> 4;
  int n0 = (w % nt) << 4;
  int bh = w / nt;
  if (bh >= BQ * NHD) return;
  const int b = bh >> 2, h = bh & 3;
  const __bf16* qb = q  + (size_t)bh * NPT * DHD;
  const __bf16* kb = kk + (size_t)bh * M * DHD;
  const __bf16* vb = vv + (size_t)bh * DHD * M;
  const int lane = threadIdx.x & 31, col = lane & 15, hs = lane >> 4;
  v16bf aq = load_a_rm(qb + (size_t)n0 * DHD, DHD);
  float rm[8], ri[8];
#pragma unroll
  for (int r = 0; r < 8; ++r) {
    int n = n0 + hs * 8 + r;
    rm[r] = rowmax[(size_t)bh * NPT + n];
    ri[r] = 1.0f / rowsum[(size_t)bh * NPT + n];
  }
  v8f macc[2] = {};
  float* prow = pbuf + (size_t)w * M;   // per-wave colsum partials (deterministic reduce later)
  for (int m0 = 0; m0 < M; m0 += 32) {
#pragma unroll
    for (int t2 = 0; t2 < 2; ++t2) {
      const int mt = m0 + 16 * t2;
      v16bf bk = load_b_kc(kb + (size_t)mt * DHD, DHD);
      v8f s = {};
      s = wmma_bf(aq, bk, s);
      float cs = 0.f;
#pragma unroll
      for (int r = 0; r < 8; ++r) {
        float p = __expf(s[r] * scale - rm[r]) * ri[r];
        cs += p;
        pl[(hs * 8 + r) * 32 + t2 * 16 + col] = (__bf16)p;   // stage prob tile in LDS
      }
      cs += __shfl_xor(cs, 16, 32);     // combine both row halves of column `col`
      if (hs == 0) prow[mt + col] = cs; // sum over the 16 queries of this tile
    }
    // wave-internal LDS RAW fence (DS ops in-order per wave; also a compiler barrier)
    asm volatile("s_wait_dscnt 0x0" ::: "memory");
    v16bf ap = load_a_rm(pl, 32);       // A[n][m] from LDS, 2x ds_load_b128 per lane
#pragma unroll
    for (int j = 0; j < 2; ++j) {
      v16bf bv = load_b_kc(vb + (size_t)(16 * j) * M + m0, M);  // B[m][d] = v^T
      macc[j] = wmma_bf(ap, bv, macc[j]);
    }
  }
  // msg point-major [B,N,128]: channel c = d*H + h (reference reshape)
#pragma unroll
  for (int j = 0; j < 2; ++j)
#pragma unroll
    for (int r = 0; r < 8; ++r) {
      int d = j * 16 + col;
      int c = d * NHD + h;
      int n = n0 + hs * 8 + r;
      msgt[((size_t)b * NPT + n) * DCH + c] = (__bf16)macc[j][r];
    }
}

// deterministic reduce of colsum partials: p[b,m] = mean_h sum_n prob
__global__ void k_colsum_reduce(const float* __restrict__ pbuf, float* __restrict__ pcol, int M) {
  int t = blockIdx.x * blockDim.x + threadIdx.x;
  if (t >= BQ * M) return;
  int m = t % M, b = t / M;
  float s = 0.f;
  const int rows = NHD * (NPT >> 4);  // 512 waves per batch
  for (int r = 0; r < rows; ++r) s += pbuf[((size_t)(b * rows + r)) * M + m];
  pcol[b * NPT + m] = s * (1.0f / NHD);
}

// ---------------- InstanceNorm1d(eps=1e-3, no affine) + ReLU -> bf16 (point-major) ----------------
__global__ void k_instnorm_relu(const float* __restrict__ t, __bf16* __restrict__ ht) {
  const int bc = blockIdx.x;                 // over B*256 rows (channel-major input)
  const int b = bc / 256, c = bc % 256;
  const float* row = t + (size_t)bc * NPT;
  __shared__ float s1[256], s2[256];
  float a = 0.f, d = 0.f;
  for (int n = threadIdx.x; n < NPT; n += blockDim.x) {
    float x = row[n]; a += x; d += x * x;
  }
  s1[threadIdx.x] = a; s2[threadIdx.x] = d;
  __syncthreads();
  for (int off = 128; off > 0; off >>= 1) {
    if ((int)threadIdx.x < off) {
      s1[threadIdx.x] += s1[threadIdx.x + off];
      s2[threadIdx.x] += s2[threadIdx.x + off];
    }
    __syncthreads();
  }
  const float mean = s1[0] * (1.0f / NPT);
  const float var  = s2[0] * (1.0f / NPT) - mean * mean;
  const float rs   = rsqrtf(var + 1e-3f);
  for (int n = threadIdx.x; n < NPT; n += blockDim.x) {
    float v = (row[n] - mean) * rs;
    ht[((size_t)b * NPT + n) * 256 + c] = (__bf16)(v > 0.f ? v : 0.f);
  }
}

// ---------------- pooling_by_prob: top-k (lower index wins ties) + sort + gather ----------------
__global__ void k_pool(const float* __restrict__ pcol, const int* __restrict__ idx_in,
                       int* __restrict__ idx_out, int M, int K) {
  const int b = blockIdx.x;
  __shared__ float pv[NPT];
  __shared__ unsigned char sel[NPT];
  const float* p = pcol + (size_t)b * NPT;
  for (int m = threadIdx.x; m < M; m += blockDim.x) pv[m] = p[m];
  __syncthreads();
  for (int m = threadIdx.x; m < M; m += blockDim.x) {
    const float pm = pv[m];
    int rank = 0;
    for (int j = 0; j < M; ++j)
      rank += (pv[j] > pm) || (pv[j] == pm && j < m);
    sel[m] = (rank < K) ? 1 : 0;
  }
  __syncthreads();
  for (int m = threadIdx.x; m < M; m += blockDim.x) {
    if (sel[m]) {
      int pos = 0;
      for (int j = 0; j < m; ++j) pos += sel[j];
      idx_out[b * NPT + pos] = idx_in[b * NPT + m];
    }
  }
}

// ---------------- host orchestration ----------------
static inline int cdiv(int a, int b) { return (a + b - 1) / b; }
static inline int wblk(int waves) { return cdiv(waves, 8); }  // 8 waves / 256-thread block

extern "C" void kernel_launch(void* const* d_in, const int* in_sizes, int n_in,
                              void* d_out, int out_size, void* d_ws, size_t ws_size,
                              hipStream_t stream) {
  (void)in_sizes; (void)n_in; (void)out_size; (void)ws_size;
  const float* in_x0 = (const float*)d_in[0];
  const float* in_x1 = (const float*)d_in[1];
  const float* in_Wq = (const float*)d_in[2];
  const float* in_bq = (const float*)d_in[3];
  const float* in_Wk = (const float*)d_in[4];
  const float* in_bk = (const float*)d_in[5];
  const float* in_Wv = (const float*)d_in[6];
  const float* in_bv = (const float*)d_in[7];
  const float* in_Wm = (const float*)d_in[8];
  const float* in_bm = (const float*)d_in[9];
  const float* in_W1 = (const float*)d_in[10];
  const float* in_b1 = (const float*)d_in[11];
  const float* in_W2 = (const float*)d_in[12];
  const float* in_b2 = (const float*)d_in[13];

  // ----- workspace carve -----
  char* wp = (char*)d_ws;
  auto carve = [&](size_t bytes) -> char* {
    char* p = wp; wp += (bytes + 255) & ~(size_t)255; return p;
  };
  const int XN = BQ * DCH * NPT;          // 524288
  float*  xf0   = (float*) carve((size_t)XN * 4);            // [B,128,N] f32 (channel-major)
  float*  xf1   = (float*) carve((size_t)XN * 4);
  __bf16* xt0   = (__bf16*)carve((size_t)XN * 2);            // [B,N,128] bf16 (point-major)
  __bf16* xt1   = (__bf16*)carve((size_t)XN * 2);
  __bf16* wq_b  = (__bf16*)carve((size_t)65536 * 2);
  __bf16* wk_b  = (__bf16*)carve((size_t)65536 * 2);
  __bf16* wv_b  = (__bf16*)carve((size_t)65536 * 2);
  __bf16* wm_b  = (__bf16*)carve((size_t)65536 * 2);
  __bf16* w1_b  = (__bf16*)carve((size_t)262144 * 2);
  __bf16* w2_b  = (__bf16*)carve((size_t)131072 * 2);
  __bf16* qb    = (__bf16*)carve((size_t)XN * 2);            // [B,H,N,32]
  __bf16* kbuf  = (__bf16*)carve((size_t)XN * 2);            // [B,H,M,32]
  __bf16* vbuf  = (__bf16*)carve((size_t)XN * 2);            // [B,H,32,M]
  __bf16* sb0   = (__bf16*)carve((size_t)XN * 2);            // [B,M,128] point-major
  __bf16* sb1   = (__bf16*)carve((size_t)XN * 2);
  __bf16* msgt  = (__bf16*)carve((size_t)XN * 2);            // [B,N,128] point-major
  __bf16* ybf   = (__bf16*)carve((size_t)BQ * 256 * NPT * 2);// [B,N,256] point-major
  float*  tb    = (float*) carve((size_t)BQ * 256 * NPT * 4);// [B,256,N] channel-major
  __bf16* hb    = (__bf16*)carve((size_t)BQ * 256 * NPT * 2);// [B,N,256] point-major
  float*  rowmx = (float*) carve((size_t)BQ * NHD * NPT * 4);
  float*  rowsm = (float*) carve((size_t)BQ * NHD * NPT * 4);
  float*  pbuf  = (float*) carve((size_t)BQ * NHD * (NPT / 16) * NPT * 4); // 8MB
  float*  pc0   = (float*) carve((size_t)BQ * NPT * 4);
  float*  pc1   = (float*) carve((size_t)BQ * NPT * 4);
  int*    idxA0 = (int*)   carve((size_t)BQ * NPT * 4);
  int*    idxB0 = (int*)   carve((size_t)BQ * NPT * 4);
  int*    idxA1 = (int*)   carve((size_t)BQ * NPT * 4);
  int*    idxB1 = (int*)   carve((size_t)BQ * NPT * 4);

  // ----- setup: copies, bf16 conversions (x transposed), iota indices -----
  k_copy_f32<<<cdiv(XN, 256), 256, 0, stream>>>(in_x0, xf0, XN);
  k_copy_f32<<<cdiv(XN, 256), 256, 0, stream>>>(in_x1, xf1, XN);
  k_cvt_xpose<<<cdiv(XN, 256), 256, 0, stream>>>(in_x0, xt0);
  k_cvt_xpose<<<cdiv(XN, 256), 256, 0, stream>>>(in_x1, xt1);
  k_cvt_bf16<<<cdiv(65536, 256), 256, 0, stream>>>(in_Wq, wq_b, 65536);
  k_cvt_bf16<<<cdiv(65536, 256), 256, 0, stream>>>(in_Wk, wk_b, 65536);
  k_cvt_bf16<<<cdiv(65536, 256), 256, 0, stream>>>(in_Wv, wv_b, 65536);
  k_cvt_bf16<<<cdiv(65536, 256), 256, 0, stream>>>(in_Wm, wm_b, 65536);
  k_cvt_bf16<<<cdiv(262144, 256), 256, 0, stream>>>(in_W1, w1_b, 262144);
  k_cvt_bf16<<<cdiv(131072, 256), 256, 0, stream>>>(in_W2, w2_b, 131072);
  k_iota_idx<<<cdiv(BQ * NPT, 256), 256, 0, stream>>>(idxA0);
  k_iota_idx<<<cdiv(BQ * NPT, 256), 256, 0, stream>>>(idxA1);

  const float scale = 0.17677669529663687f; // 32^-0.5
  const int Ms[4] = {2048, 1024, 512, 256};
  const bool self_l[4] = {true, false, true, false};
  int *cur0 = idxA0, *cur1 = idxA1, *nxt0 = idxB0, *nxt1 = idxB1;

  for (int l = 0; l < 4; ++l) {
    const int M = Ms[l], Kp = M / 2;   // max(M/2,128) == M/2 for all layers here
    const bool self = self_l[l];
    const __bf16* Wq_l = wq_b + (size_t)l * 16384;
    const __bf16* Wk_l = wk_b + (size_t)l * 16384;
    const __bf16* Wv_l = wv_b + (size_t)l * 16384;
    const __bf16* Wm_l = wm_b + (size_t)l * 16384;
    const __bf16* W1_l = w1_b + (size_t)l * 65536;
    const __bf16* W2_l = w2_b + (size_t)l * 32768;
    const float* bq_l = in_bq + l * 128;
    const float* bk_l = in_bk + l * 128;
    const float* bv_l = in_bv + l * 128;
    const float* bm_l = in_bm + l * 128;
    const float* b1_l = in_b1 + l * 256;
    const float* b2_l = in_b2 + l * 128;

    // gather both sources BEFORE either stream updates x (matches reference ordering)
    const int gth = BQ * M * 16;
    if (self) {
      k_gather_rows<<<cdiv(gth, 256), 256, 0, stream>>>(xt0, cur0, sb0, M);
      k_gather_rows<<<cdiv(gth, 256), 256, 0, stream>>>(xt1, cur1, sb1, M);
    } else {
      k_gather_rows<<<cdiv(gth, 256), 256, 0, stream>>>(xt1, cur1, sb0, M);
      k_gather_rows<<<cdiv(gth, 256), 256, 0, stream>>>(xt0, cur0, sb1, M);
    }

    for (int t = 0; t < 2; ++t) {
      float*  xf = t ? xf1 : xf0;
      __bf16* xt = t ? xt1 : xt0;
      __bf16* sb = t ? sb1 : sb0;
      float*  pc = t ? pc1 : pc0;

      // q/k/v projections (WMMA GEMMs): q,k point-major; v channel-major
      k_gemm_qkv<<<wblk(BQ * 8 * (NPT / 64)), 256, 0, stream>>>(Wq_l, bq_l, xt, qb, NPT, 1);
      k_gemm_qkv<<<wblk(BQ * 8 * (M / 64)),   256, 0, stream>>>(Wk_l, bk_l, sb, kbuf, M, 1);
      k_gemm_qkv<<<wblk(BQ * 8 * (M / 64)),   256, 0, stream>>>(Wv_l, bv_l, sb, vbuf, M, 0);

      // flash-style attention: stats pass then apply pass (score GEMM recomputed)
      k_attn_stats<<<wblk(BQ * NHD * (NPT / 16)), 256, 0, stream>>>(qb, kbuf, rowmx, rowsm, M, scale);
      k_attn_apply<<<wblk(BQ * NHD * (NPT / 16)), 256, 0, stream>>>(qb, kbuf, vbuf, rowmx, rowsm,
                                                                    pbuf, msgt, M, scale);
      k_colsum_reduce<<<cdiv(BQ * M, 256), 256, 0, stream>>>(pbuf, pc, M);

      // Wm(msg) -> channels [128,256) of y ; copy x -> channels [0,128)   (both point-major)
      k_gemm_bf16t<<<wblk(BQ * 8 * (NPT / 64)), 256, 0, stream>>>(Wm_l, bm_l, msgt, ybf + 128,
                                                                  DCH, DCH, NPT, 256, 256 * NPT);
      k_x_to_y<<<cdiv(BQ * NPT * 16, 256), 256, 0, stream>>>(xt, ybf);

      // MLP: t = W1*y + b1 ; h = relu(instnorm(t)) ; x += W2*h + b2
      k_gemm_f32<<<wblk(BQ * 16 * (NPT / 64)), 256, 0, stream>>>(W1_l, b1_l, ybf, tb, 256, 256, NPT);
      k_instnorm_relu<<<BQ * 256, 256, 0, stream>>>(tb, hb);
      k_gemm_resid<<<wblk(BQ * 8 * (NPT / 64)), 256, 0, stream>>>(W2_l, b2_l, hb, xf, xt, NPT);
    }

    // pooling: prob of a stream pools the idx that fed its source
    if (self) {
      k_pool<<<BQ, 256, 0, stream>>>(pc0, cur0, nxt0, M, Kp);
      k_pool<<<BQ, 256, 0, stream>>>(pc1, cur1, nxt1, M, Kp);
    } else {
      k_pool<<<BQ, 256, 0, stream>>>(pc0, cur1, nxt1, M, Kp);
      k_pool<<<BQ, 256, 0, stream>>>(pc1, cur0, nxt0, M, Kp);
    }
    int* tmp;
    tmp = cur0; cur0 = nxt0; nxt0 = tmp;
    tmp = cur1; cur1 = nxt1; nxt1 = tmp;
  }

  // output: stack([x0, x1]) -> [2,B,D,N] f32
  float* out = (float*)d_out;
  k_copy_f32<<<cdiv(XN, 256), 256, 0, stream>>>(xf0, out, XN);
  k_copy_f32<<<cdiv(XN, 256), 256, 0, stream>>>(xf1, out + XN, XN);
}
